// st_dmrgcn_48996986913316
// MI455X (gfx1250) — compile-verified, matrix operand order checked by reading, not run on hardware
//
#include <hip/hip_runtime.h>

// ST-DMR-GCN block for MI455X (gfx1250, wave32, WMMA f16 16x16x32).
// Pass 1: per-(n,t) block: async-copy A slice to LDS (global_load_async_to_lds_b128),
//         build L (normalized Laplacian of disentangled A) and H (1x1 conv) in LDS
//         WMMA layouts, accumulate 6 graph matmuls with v_wmma_f32_16x16x32_f16,
//         PReLU, store f16 y to workspace.
// Pass 2: temporal 3-tap conv as 64x192 x 192x128 WMMA matmul + bias +
//         residual + PReLU -> f32 output.

typedef _Float16 v16h __attribute__((ext_vector_type(16)));
typedef _Float16 v2h  __attribute__((ext_vector_type(2)));
typedef float    v8f  __attribute__((ext_vector_type(8)));

// Position (in 16-bit units) of element (k_local, m_or_n_local) inside one
// 16x16(MN) x 32(K) WMMA f16 operand tile stored as [lane(32)][elem(16)].
// ISA 7.12.2: lane = mn + 16*half, K = 16*(j>=4) + (j&3)*2 + half*8 + p,
// v16h element index = 2*j + p.
__device__ __forceinline__ int wpos(int kl, int mn) {
    int hi   = kl >> 4;        // 0..1
    int rem  = kl & 15;
    int half = rem >> 3;       // 0..1
    int rem2 = rem & 7;
    int j    = (hi << 2) + (rem2 >> 1);
    int p    = rem2 & 1;
    return (mn + (half << 4)) * 16 + ((j << 1) + p);
}

__device__ __forceinline__ v8f wmma16(v16h a, v16h b, v8f c) {
    return __builtin_amdgcn_wmma_f32_16x16x32_f16(
        /*neg_a=*/false, a, /*neg_b=*/false, b,
        /*c_mod=*/(short)0, c, /*reuse_a=*/false, /*reuse_b=*/false);
}

// CDNA5 async memory->LDS copy (ASYNCcnt-tracked, no VGPR round-trip).
// GVS mode: mem = SADDR64 + VADDR32 + ioffset; LDS dst = wave LDS base + VDST32.
__device__ __forceinline__ void async_b128(unsigned lds_off, unsigned gvo,
                                           const void* sbase) {
    asm volatile("global_load_async_to_lds_b128 %0, %1, %2"
                 :: "v"(lds_off), "v"(gvo), "s"(sbase) : "memory");
}
__device__ __forceinline__ void wait_async0() {
    asm volatile("s_wait_asynccnt 0x0" ::: "memory");
}

// ---------------- Pass 1: GCN branches ----------------
// LDS carve (bytes):
//   sA    :      0 .. 65536   A slice 128x128 f32
//   sX    :  65536 .. 81920   X  B-layout  (2 ktile x 8 ntile) f16
//   sW    :  81920 .. 90112   Ws A-layout  (4 mtile x 2 ktile) f16
//   sH    :  90112 ..106496   H  A-layout  (4 mtile x 4 ktile) f16
//   sL    : 106496 ..139264   L  B-layout  (4 ktile x 8 ntile) f16
//   sDinv : 139264 ..140800   3 x 128 f32
//   sDeg  : 140800 ..143872   3 x 256 f32
#define P1_SMEM 143872

__global__ void __launch_bounds__(256)
gcn_pass1(const float* __restrict__ x, const float* __restrict__ A,
          const float* __restrict__ w0, const float* __restrict__ b0,
          const float* __restrict__ w1, const float* __restrict__ b1,
          const float* __restrict__ prelu_t, _Float16* __restrict__ y_ws) {
    extern __shared__ unsigned char smem[];
    float*    sA    = (float*)smem;
    _Float16* sX    = (_Float16*)(smem + 65536);
    _Float16* sW    = (_Float16*)(smem + 81920);
    _Float16* sH    = (_Float16*)(smem + 90112);
    _Float16* sL    = (_Float16*)(smem + 106496);
    float*    sDinv = (float*)(smem + 139264);
    float*    sDeg  = (float*)(smem + 140800);

    const int tid   = threadIdx.x;
    const int wv    = tid >> 5;
    const int lane  = tid & 31;
    const int nl    = lane & 15;
    const int halfD = lane >> 4;
    const int n     = blockIdx.x >> 3;
    const int t     = blockIdx.x & 7;

    // Kick off async A staging for rel=0 immediately (overlaps with X staging).
    const unsigned ldsA = (unsigned)(uintptr_t)sA;  // wave-relative LDS offset
    {
        const void* Abase = (const void*)(A + ((((size_t)n * 2 + 0) * 8 + t) << 14));
#pragma unroll
        for (int i = 0; i < 16; ++i) {
            unsigned off = (unsigned)(tid * 16 + i * 4096);
            async_b128(ldsA + off, off, Abase);
        }
        // Prefetch the rel=1 slice into L2 while rel=0 is processed.
        const char* Anext = (const char*)(A + ((((size_t)n * 2 + 1) * 8 + t) << 14));
        __builtin_prefetch(Anext + tid * 256, 0, 3);
        __builtin_prefetch(Anext + tid * 256 + 128, 0, 3);
    }

    // Stage X (B-layout, k = c_in, n = v) once: x[n, c, t, v].
    {
        int v = tid & 127, hc = tid >> 7;
        const float* xp = x + (((size_t)n * 64 * 8 + t) * 128) + v;
#pragma unroll
        for (int c0 = hc * 32; c0 < hc * 32 + 32; c0 += 2) {
            float f0 = xp[(size_t)c0 * 1024];
            float f1 = xp[(size_t)(c0 + 1) * 1024];
            int tile = ((c0 >> 5) << 3) + (v >> 4);
            v2h pr = {(_Float16)f0, (_Float16)f1};
            *(v2h*)(sX + tile * 512 + wpos(c0 & 31, v & 15)) = pr;
        }
    }

    v8f acc[4] = {};
    const int ct  = wv & 3;
    const int wtb = (wv >> 2) << 2;

    for (int rel = 0; rel < 2; ++rel) {
        const float* wg = rel ? w1 : w0;
        const float* bg = rel ? b1 : b0;
        if (rel == 1) {
            __syncthreads();  // everyone done reading rel=0 sA
            const void* Abase = (const void*)(A + ((((size_t)n * 2 + 1) * 8 + t) << 14));
#pragma unroll
            for (int i = 0; i < 16; ++i) {
                unsigned off = (unsigned)(tid * 16 + i * 4096);
                async_b128(ldsA + off, off, Abase);
            }
        }
        wait_async0();
        __syncthreads();
        // Degrees: per row, count mask membership (strict intervals).
        {
            int row = tid & 127, hf = tid >> 7;
            const float* ar = sA + row * 128 + hf * 64;
            float c0 = 0.f, c1 = 0.f, c2 = 0.f;
            for (int j2 = 0; j2 < 64; ++j2) {
                float a = ar[(j2 + row) & 63];  // rotate to avoid bank conflicts
                c0 += (a > 0.0f  && a < 0.25f) ? 1.0f : 0.0f;
                c1 += (a > 0.25f && a < 0.5f)  ? 1.0f : 0.0f;
                c2 += (a > 0.5f)               ? 1.0f : 0.0f;
            }
            sDeg[0 * 256 + hf * 128 + row] = c0;
            sDeg[1 * 256 + hf * 128 + row] = c1;
            sDeg[2 * 256 + hf * 128 + row] = c2;
        }
        __syncthreads();
        if (tid < 128) {
#pragma unroll
            for (int s = 0; s < 3; ++s) {
                float d = 1.0f + sDeg[s * 256 + tid] + sDeg[s * 256 + 128 + tid];
                sDinv[s * 128 + tid] = rsqrtf(d);
            }
        }
        __syncthreads();

        for (int s = 0; s < 3; ++s) {
            // Stage W_s (A-layout, m = c_out, k = c_in).
            {
                int m = tid & 63, part = tid >> 6;
                const float* wr = wg + (size_t)(s * 64 + m) * 64;
#pragma unroll
                for (int k = part * 16; k < part * 16 + 16; k += 2) {
                    int tile = ((m >> 4) << 1) + (k >> 5);
                    v2h pr = {(_Float16)wr[k], (_Float16)wr[k + 1]};
                    *(v2h*)(sW + tile * 512 + wpos(k & 31, m & 15)) = pr;
                }
            }
            __syncthreads();
            // H = W_s * X + b  (WMMA), scatter f16 into A-layout sH.
            {
                int id = wv << 2;
#pragma unroll
                for (int q = 0; q < 4; ++q) {
                    int tmt = (id + q) >> 3, tvt = (id + q) & 7;
                    v8f hacc = {};
#pragma unroll
                    for (int kk = 0; kk < 2; ++kk) {
                        v16h aop = *(const v16h*)(sW + ((tmt * 2 + kk) * 512) + lane * 16);
                        v16h bop = *(const v16h*)(sX + ((kk * 8 + tvt) * 512) + lane * 16);
                        hacc = wmma16(aop, bop, hacc);
                    }
                    int kl = ((tvt & 1) << 4) + nl;  // k_local = v & 31
                    _Float16* dst = sH + (tmt * 4 + (tvt >> 1)) * 512;
#pragma unroll
                    for (int i = 0; i < 8; ++i) {
                        int ml = i + (halfD << 3);
                        float hv = hacc[i] + bg[s * 64 + tmt * 16 + ml];
                        dst[wpos(kl, ml)] = (_Float16)hv;
                    }
                }
            }
            __syncthreads();
            // Build L_s (B-layout: k = v, n = w): (mask + I) * dinv[w]*dinv[v].
            {
                int w = tid & 127, hv = tid >> 7;
                const float* ar = sA + w * 128;
                float dw = sDinv[s * 128 + w];
#pragma unroll
                for (int vv = 0; vv < 64; vv += 2) {
                    int v0 = hv * 64 + ((vv + 2 * (w & 31)) & 63);  // even, rotated
                    float2 a2 = *(const float2*)(ar + v0);
                    float m0, m1;
                    if (s == 0) {
                        m0 = (a2.x > 0.0f && a2.x < 0.25f) ? 1.0f : 0.0f;
                        m1 = (a2.y > 0.0f && a2.y < 0.25f) ? 1.0f : 0.0f;
                    } else if (s == 1) {
                        m0 = (a2.x > 0.25f && a2.x < 0.5f) ? 1.0f : 0.0f;
                        m1 = (a2.y > 0.25f && a2.y < 0.5f) ? 1.0f : 0.0f;
                    } else {
                        m0 = (a2.x > 0.5f) ? 1.0f : 0.0f;
                        m1 = (a2.y > 0.5f) ? 1.0f : 0.0f;
                    }
                    if (w == v0)     m0 += 1.0f;
                    if (w == v0 + 1) m1 += 1.0f;
                    float l0 = m0 * dw * sDinv[s * 128 + v0];
                    float l1 = m1 * dw * sDinv[s * 128 + v0 + 1];
                    int tile = ((v0 >> 5) << 3) + (w >> 4);
                    v2h pr = {(_Float16)l0, (_Float16)l1};
                    *(v2h*)(sL + tile * 512 + wpos(v0 & 31, w & 15)) = pr;
                }
            }
            __syncthreads();
            // out[c, w] += sum_v H[c, v] * L[w, v]   (A = H, B = L^T layout).
#pragma unroll
            for (int q = 0; q < 4; ++q) {
                int wt = wtb + q;
#pragma unroll
                for (int kt = 0; kt < 4; ++kt) {
                    v16h aop = *(const v16h*)(sH + (ct * 4 + kt) * 512 + lane * 16);
                    v16h bop = *(const v16h*)(sL + (kt * 8 + wt) * 512 + lane * 16);
                    acc[q] = wmma16(aop, bop, acc[q]);
                }
            }
            __syncthreads();
        }
    }

    // Epilogue: PReLU(tcn), repack f16 into B-layout (k = c, n = v) in LDS,
    // then contiguous b128 copy to workspace.
    float pt = prelu_t[0];
    _Float16* sY = (_Float16*)smem;  // reuse sA region (all readers synced out)
#pragma unroll
    for (int q = 0; q < 4; ++q) {
        int wt = wtb + q;
#pragma unroll
        for (int i = 0; i < 8; ++i) {
            int c = ct * 16 + i + (halfD << 3);
            float r = acc[q][i];
            r = r > 0.0f ? r : pt * r;
            int tile = ((c >> 5) << 3) + wt;
            sY[tile * 512 + wpos(c & 31, nl)] = (_Float16)r;
        }
    }
    __syncthreads();
    {
        const uint4* src = (const uint4*)sY;
        uint4* dst = (uint4*)(y_ws + ((size_t)(n * 8 + t) * 8192));
        for (int i = tid; i < 1024; i += 256) dst[i] = src[i];
    }
}

// ---------------- Pass 2: temporal conv + residual ----------------
// LDS: sWt = wtcn A-layout (4 mtile x 6 ktile) f16 = 24576 bytes.
#define P2_SMEM 24576

__global__ void __launch_bounds__(256)
tcn_pass2(const _Float16* __restrict__ y_ws, const float* __restrict__ w_tcn,
          const float* __restrict__ b_tcn, const float* __restrict__ x,
          const float* __restrict__ prelu_o, float* __restrict__ out) {
    extern __shared__ unsigned char smem[];
    _Float16* sWt = (_Float16*)smem;

    const int tid   = threadIdx.x;
    const int wv    = tid >> 5;
    const int lane  = tid & 31;
    const int nl    = lane & 15;
    const int halfD = lane >> 4;
    const int n     = blockIdx.x >> 3;
    const int t     = blockIdx.x & 7;

    // Stage wtcn (A-layout, m = c_out, k = dt*64 + c_in), from (O, I, 3, 1).
    {
        int m = tid & 63, grp = tid >> 6;
        for (int kk = grp * 24; kk < grp * 24 + 24; ++kk) {
            int k = kk << 1;           // even k; k,k+1 share dt
            int dt = k >> 6, ci = k & 63;
            float f0 = w_tcn[(size_t)(m * 64 + ci) * 3 + dt];
            float f1 = w_tcn[(size_t)(m * 64 + ci + 1) * 3 + dt];
            int tile = (m >> 4) * 6 + (k >> 5);
            v2h pr = {(_Float16)f0, (_Float16)f1};
            *(v2h*)(sWt + tile * 512 + wpos(k & 31, m & 15)) = pr;
        }
    }
    __syncthreads();

    const int ct  = wv & 3;
    const int wtb = (wv >> 2) << 2;
    v8f acc[4] = {};

#pragma unroll
    for (int kg = 0; kg < 6; ++kg) {
        int dt = kg >> 1, ktl = kg & 1;
        int tp = t + dt - 1;
        if (tp < 0 || tp >= 8) continue;  // zero padding -> skip (uniform branch)
        v16h aop = *(const v16h*)(sWt + (ct * 6 + kg) * 512 + lane * 16);
        const _Float16* yb = y_ws + ((size_t)(n * 8 + tp) * 16 + ktl * 8) * 512;
#pragma unroll
        for (int q = 0; q < 4; ++q) {
            v16h bop = *(const v16h*)(yb + (wtb + q) * 512 + lane * 16);
            acc[q] = wmma16(aop, bop, acc[q]);
        }
    }

    float po = prelu_o[0];
#pragma unroll
    for (int q = 0; q < 4; ++q) {
        int wt = wtb + q;
#pragma unroll
        for (int i = 0; i < 8; ++i) {
            int c = ct * 16 + i + (halfD << 3);
            int v = wt * 16 + nl;
            size_t idx = (((size_t)n * 64 + c) * 8 + t) * 128 + v;
            float r = acc[q][i] + b_tcn[c] + x[idx];
            out[idx] = r > 0.0f ? r : po * r;
        }
    }
}

extern "C" void kernel_launch(void* const* d_in, const int* in_sizes, int n_in,
                              void* d_out, int out_size, void* d_ws, size_t ws_size,
                              hipStream_t stream) {
    (void)in_sizes; (void)n_in; (void)out_size; (void)ws_size;
    const float* x  = (const float*)d_in[0];
    const float* A  = (const float*)d_in[1];
    const float* w0 = (const float*)d_in[2];
    const float* b0 = (const float*)d_in[3];
    const float* w1 = (const float*)d_in[4];
    const float* b1 = (const float*)d_in[5];
    const float* wt = (const float*)d_in[6];
    const float* bt = (const float*)d_in[7];
    const float* pt = (const float*)d_in[8];
    const float* po = (const float*)d_in[9];
    _Float16* y_ws  = (_Float16*)d_ws;   // needs 128*8*64*128*2 = 16 MiB

    dim3 grid(128 * 8), block(256);
    gcn_pass1<<<grid, block, P1_SMEM, stream>>>(x, A, w0, b0, w1, b1, pt, y_ws);
    tcn_pass2<<<grid, block, P2_SMEM, stream>>>(y_ws, wt, bt, x, po, (float*)d_out);
}
